// EncoderWithMemory_38620345926137
// MI455X (gfx1250) — matrix operand order repaired and checked by reading, not compiled, not run on hardware
//
#include <hip/hip_runtime.h>
#include <hip/hip_bf16.h>
#include <stdint.h>

// ---------------- problem constants (from reference) ----------------
#define B_    512
#define IN_   2048
#define HD_   1024
#define MS_   1024
#define MD_   64
#define CD_   128
#define CH_   1156        // 2*MD + MS + 4
#define CH4_  4624        // 4*CH
#define HD4_  4096        // 4*HD
#define EPS_  1e-8f

typedef __attribute__((ext_vector_type(16))) __bf16   v16bf;
typedef __attribute__((ext_vector_type(8)))  float    v8f;
typedef __attribute__((ext_vector_type(8)))  uint32_t u32x8;
typedef __attribute__((ext_vector_type(4)))  uint32_t u32x4;
typedef __attribute__((ext_vector_type(4)))  int32_t  i32x4;
typedef __attribute__((ext_vector_type(8)))  int32_t  i32x8;

// One-instruction f32x2 -> packed bf16 conversion.
#if __has_builtin(__builtin_amdgcn_cvt_pk_bf16_f32)
typedef __attribute__((ext_vector_type(2))) __bf16 v2bf;
__device__ __forceinline__ uint32_t pack2_bf16(float lo, float hi) {
  v2bf r = __builtin_amdgcn_cvt_pk_bf16_f32(lo, hi);   // v_cvt_pk_bf16_f32 (RNE)
  return __builtin_bit_cast(uint32_t, r);
}
#else
__device__ __forceinline__ uint32_t pack2_bf16(float lo, float hi) {
  // v_perm_b32: take high 16 bits of each float (truncation round)
  return __builtin_amdgcn_perm(__builtin_bit_cast(uint32_t, hi),
                               __builtin_bit_cast(uint32_t, lo),
                               0x07060302u);
}
#endif

__device__ __forceinline__ float sigm(float x) { return 1.0f / (1.0f + __expf(-x)); }
__device__ __forceinline__ int   imin(int a, int b) { return a < b ? a : b; }

// Convert 4 x float4 (16 consecutive f32) into one 16-wide bf16 fragment.
__device__ __forceinline__ v16bf pack_frag(const float4 r[4]) {
  u32x8 u;
  u[0] = pack2_bf16(r[0].x, r[0].y); u[1] = pack2_bf16(r[0].z, r[0].w);
  u[2] = pack2_bf16(r[1].x, r[1].y); u[3] = pack2_bf16(r[1].z, r[1].w);
  u[4] = pack2_bf16(r[2].x, r[2].y); u[5] = pack2_bf16(r[2].z, r[2].w);
  u[6] = pack2_bf16(r[3].x, r[3].y); u[7] = pack2_bf16(r[3].z, r[3].w);
  return __builtin_bit_cast(v16bf, u);
}

// ====================================================================
// GEMM k-step body: phase 1 = issue ALL loads (A + NT B-tiles),
// phase 2 = convert, phase 3 = back-to-back WMMAs.
// ====================================================================
template<int NT>
__device__ __forceinline__ void gemm_kstep(const float* Arow,
                                           const float* const Wrow[NT],
                                           int kb, int half, v8f acc[NT])
{
  float4 ar[4];
  float4 br[NT][4];
  ar[0] = *(const float4*)(Arow + kb + half * 16);
  ar[1] = *(const float4*)(Arow + kb + half * 16 + 4);
  ar[2] = *(const float4*)(Arow + kb + half * 16 + 8);
  ar[3] = *(const float4*)(Arow + kb + half * 16 + 12);
  #pragma unroll
  for (int t = 0; t < NT; ++t) {
    const float* wp = Wrow[t] + kb + half * 16;
    br[t][0] = *(const float4*)(wp);
    br[t][1] = *(const float4*)(wp + 4);
    br[t][2] = *(const float4*)(wp + 8);
    br[t][3] = *(const float4*)(wp + 12);
  }
  v16bf afrag = pack_frag(ar);
  v16bf bfrag[NT];
  #pragma unroll
  for (int t = 0; t < NT; ++t) bfrag[t] = pack_frag(br[t]);
  #pragma unroll
  for (int t = 0; t < NT; ++t)
    acc[t] = __builtin_amdgcn_wmma_f32_16x16x32_bf16(
        false, afrag, false, bfrag[t], (short)0, acc[t], false, false);
}

// ====================================================================
// GEMM: C[M,N] = (beta ? C : 0) + A[M,K](lda) @ W[N,K]^T(ldw) + bias[N]
// One wave computes a 16x(16*NT) output strip with bf16 WMMA, f32 acc.
// ====================================================================
template<int NT>
__global__ void __launch_bounds__(256)
gemm_wmma_bf16(const float* __restrict__ A, int lda,
               const float* __restrict__ W, int ldw,
               const float* __restrict__ bias,
               float* __restrict__ C, int M, int N, int K, int beta)
{
  const int wave   = (blockIdx.x * 256 + threadIdx.x) >> 5;
  const int lane   = threadIdx.x & 31;
  const int tilesN = (N + 15) >> 4;
  const int strips = (tilesN + NT - 1) / NT;
  const int tm     = wave / strips;
  const int st     = wave % strips;
  if (tm * 16 >= M) return;               // wave-uniform exit
  const int tn0  = st * NT;
  const int l16  = lane & 15;
  const int half = lane >> 4;

  const float* Arow = A + (long)(tm * 16 + l16) * lda;

  // Clamp W row pointers so the hot loop has no N-edge branches.
  const float* Wrow[NT];
  #pragma unroll
  for (int t = 0; t < NT; ++t) {
    int ncol = (tn0 + t) * 16 + l16;
    Wrow[t] = W + (long)imin(ncol, N - 1) * ldw;
  }

  v8f acc[NT];
  const v8f vzero = {0.f,0.f,0.f,0.f,0.f,0.f,0.f,0.f};
  #pragma unroll
  for (int t = 0; t < NT; ++t) acc[t] = vzero;

  const int Kmain = K & ~31;
  #pragma unroll 2
  for (int kb = 0; kb < Kmain; kb += 32)
    gemm_kstep<NT>(Arow, Wrow, kb, half, acc);

  // ---- single guarded tail iteration (only when K % 32 != 0) ----
  if (K > Kmain) {
    const int kb = Kmain;
    u32x8 au;
    const int ka0 = kb + half * 16;
    const int ka1 = kb + half * 16 + 8;
    #pragma unroll
    for (int j = 0; j < 4; ++j) {
      int k0 = ka0 + 2 * j;
      float x0 = (k0     < K) ? Arow[k0]     : 0.0f;
      float x1 = (k0 + 1 < K) ? Arow[k0 + 1] : 0.0f;
      au[j] = pack2_bf16(x0, x1);
      int k1 = ka1 + 2 * j;
      float y0 = (k1     < K) ? Arow[k1]     : 0.0f;
      float y1 = (k1 + 1 < K) ? Arow[k1 + 1] : 0.0f;
      au[4 + j] = pack2_bf16(y0, y1);
    }
    v16bf afrag = __builtin_bit_cast(v16bf, au);
    #pragma unroll
    for (int t = 0; t < NT; ++t) {
      u32x8 bu;
      const int kb0 = kb + half * 16;
      #pragma unroll
      for (int j = 0; j < 8; ++j) {
        int k = kb0 + 2 * j;
        float x0 = (k     < K) ? Wrow[t][k]     : 0.0f;
        float x1 = (k + 1 < K) ? Wrow[t][k + 1] : 0.0f;
        bu[j] = pack2_bf16(x0, x1);
      }
      v16bf bfrag = __builtin_bit_cast(v16bf, bu);
      acc[t] = __builtin_amdgcn_wmma_f32_16x16x32_bf16(
          false, afrag, false, bfrag, (short)0, acc[t], false, false);
    }
  }

  // ---- store (C layout: VGPR r -> row r+8*half, lane%16 -> col) ----
  #pragma unroll
  for (int t = 0; t < NT; ++t) {
    if ((tn0 + t) * 16 >= N) continue;    // wave-uniform
    const int col = (tn0 + t) * 16 + l16;
    const float bv = bias ? bias[col] : 0.0f;
    #pragma unroll
    for (int r = 0; r < 8; ++r) {
      const int row = tm * 16 + half * 8 + r;
      long idx = (long)row * N + col;
      float v = acc[t][r] + bv;
      if (beta) v += C[idx];
      C[idx] = v;
    }
  }
}

// ====================================================================
// Split-K GEMM (for huge-K skinny-N cases): each wave handles one
// (m-tile, n-strip, k-chunk); raw partials to Cpart[sk][M][N].
// Requires K % 32 == 0 and kc % 32 == 0.
// ====================================================================
template<int NT>
__global__ void __launch_bounds__(256)
gemm_wmma_bf16_splitk(const float* __restrict__ A, int lda,
                      const float* __restrict__ W, int ldw,
                      float* __restrict__ Cpart,
                      int M, int N, int K, int SK, int kc)
{
  const int wave   = (blockIdx.x * 256 + threadIdx.x) >> 5;
  const int lane   = threadIdx.x & 31;
  const int tilesN = (N + 15) >> 4;
  const int strips = (tilesN + NT - 1) / NT;
  const int tm     = wave / (strips * SK);
  if (tm * 16 >= M) return;               // wave-uniform exit
  const int rem  = wave % (strips * SK);
  const int st   = rem / SK;
  const int sk   = rem % SK;
  const int tn0  = st * NT;
  const int l16  = lane & 15;
  const int half = lane >> 4;
  const int k0   = sk * kc;
  const int k1   = imin(K, k0 + kc);

  const float* Arow = A + (long)(tm * 16 + l16) * lda;
  const float* Wrow[NT];
  #pragma unroll
  for (int t = 0; t < NT; ++t) {
    int ncol = (tn0 + t) * 16 + l16;
    Wrow[t] = W + (long)imin(ncol, N - 1) * ldw;
  }

  v8f acc[NT];
  const v8f vzero = {0.f,0.f,0.f,0.f,0.f,0.f,0.f,0.f};
  #pragma unroll
  for (int t = 0; t < NT; ++t) acc[t] = vzero;

  #pragma unroll 1
  for (int kb = k0; kb < k1; kb += 32)
    gemm_kstep<NT>(Arow, Wrow, kb, half, acc);

  float* Cp = Cpart + (long)sk * M * N;
  #pragma unroll
  for (int t = 0; t < NT; ++t) {
    if ((tn0 + t) * 16 >= N) continue;
    const int col = (tn0 + t) * 16 + l16;
    #pragma unroll
    for (int r = 0; r < 8; ++r) {
      const int row = tm * 16 + half * 8 + r;
      Cp[(long)row * N + col] = acc[t][r];
    }
  }
}

// Deterministic fixed-order reduction of split-K partials onto C.
__global__ void splitk_reduce_kernel(float* __restrict__ C,
                                     const float* __restrict__ part,
                                     int SK, long MN)
{
  long i = (long)blockIdx.x * blockDim.x + threadIdx.x;
  if (i >= MN) return;
  float s = C[i];
  for (int k = 0; k < SK; ++k) s += part[(long)k * MN + i];
  C[i] = s;
}

// ====================================================================
// TDM prefetch: pull a weight matrix through L2 with the Tensor Data
// Mover (64x64 f32 tiles -> LDS, data discarded). One wave per tile.
// ====================================================================
__global__ void __launch_bounds__(32)
tdm_prefetch_kernel(const float* __restrict__ base, int rows, int cols, int ld)
{
  __shared__ float stage[64 * 64];
  (void)stage;
  const int tilesX = (cols + 63) >> 6;
  const int tx = blockIdx.x % tilesX;
  const int ty = blockIdx.x / tilesX;
  const int r0 = ty * 64, c0 = tx * 64;
  if (r0 >= rows) return;
  const uint32_t th = (uint32_t)imin(64, rows - r0);
  const uint32_t tw = (uint32_t)imin(64, cols - c0);
  const uint64_t gaddr = (uint64_t)(uintptr_t)(base + (long)r0 * ld + c0);

  u32x4 g0;
  g0[0] = 1u;                                                   // count=1
  g0[1] = 0u;                                                   // lds_addr
  g0[2] = (uint32_t)(gaddr & 0xFFFFFFFFu);                      // addr[31:0]
  g0[3] = (uint32_t)((gaddr >> 32) & 0x01FFFFFFu) | (2u << 30); // addr[56:32], type=2

  const uint32_t d0 = tw, d1 = th;          // tensor dims (tile is in-bounds)
  const uint32_t s0 = (uint32_t)ld;         // dim0 stride (elements)
  i32x8 g1;
  g1[0] = (int32_t)(2u << 16);                                  // data_size=4B
  g1[1] = (int32_t)((d0 & 0xFFFFu) << 16);                      // dim0[15:0]
  g1[2] = (int32_t)(((d0 >> 16) & 0xFFFFu) | ((d1 & 0xFFFFu) << 16));
  g1[3] = (int32_t)(((d1 >> 16) & 0xFFFFu) | (tw << 16));       // tile_dim0
  g1[4] = (int32_t)(th & 0xFFFFu);                              // tile_dim1
  g1[5] = (int32_t)s0;                                          // dim0_stride[31:0]
  g1[6] = 0;
  g1[7] = 0;

  const i32x4 gz = {0, 0, 0, 0};
#if __clang_major__ >= 23
  const i32x8 gz8 = {0, 0, 0, 0, 0, 0, 0, 0};
  __builtin_amdgcn_tensor_load_to_lds(g0, g1, gz, gz, gz8, 0);
#else
  __builtin_amdgcn_tensor_load_to_lds(g0, g1, gz, gz, 0);
#endif
  __builtin_amdgcn_s_wait_tensorcnt(0);
}

// ====================================================================
// LSTM pointwise: g[M,4N] + c_in[M,N] -> h_out, c_out
// ====================================================================
__global__ void lstm_pointwise_kernel(const float* __restrict__ g,
                                      const float* __restrict__ cin,
                                      float* __restrict__ hout,
                                      float* __restrict__ cout,
                                      int M, int Nc)
{
  long i = (long)blockIdx.x * blockDim.x + threadIdx.x;
  if (i >= (long)M * Nc) return;
  int row = (int)(i / Nc), col = (int)(i % Nc);
  const float* gr = g + (long)row * 4 * Nc;
  float iv = sigm(gr[col]);
  float fv = sigm(gr[Nc + col]);
  float gv = tanhf(gr[2 * Nc + col]);
  float ov = sigm(gr[3 * Nc + col]);
  float c2 = fv * cin[i] + iv * gv;
  cout[i] = c2;
  hout[i] = ov * tanhf(c2);
}

// ====================================================================
// Per-(b,m) row stats over MD=64: sum and sum-of-squares
// ====================================================================
__global__ void rowstats_kernel(const float* __restrict__ mem,
                                float* __restrict__ msum,
                                float* __restrict__ msq)
{
  long i = (long)blockIdx.x * blockDim.x + threadIdx.x;
  if (i >= (long)B_ * MS_) return;
  const float4* p = (const float4*)(mem + i * MD_);
  float s = 0.f, q = 0.f;
  #pragma unroll
  for (int j = 0; j < MD_ / 4; ++j) {
    float4 v = p[j];
    s += v.x + v.y + v.z + v.w;
    q += v.x * v.x + v.y * v.y + v.z * v.z + v.w * v.w;
  }
  msum[i] = s;
  msq[i]  = q;
}

// ====================================================================
// Content addressing: cosine -> softmax over MS -> gate with heads.
// ====================================================================
__global__ void __launch_bounds__(256)
addressing_kernel(const float* __restrict__ hn,    // [B, CH]
                  const float* __restrict__ heads, // [B, MS]
                  const float* __restrict__ msum,
                  const float* __restrict__ msq,
                  float* __restrict__ wout)        // [B, MS]
{
  const int b   = blockIdx.x;
  const int tid = threadIdx.x;
  const float* h = hn + (long)b * CH_;
  const float kstr = __expf(h[MD_]);
  const float gate = sigm(h[MD_ + 1]);
  const float sqrtMD = 8.0f;   // sqrt(64)

  __shared__ float red[256];
  float s[4];
  float mx = -3.4e38f;
  #pragma unroll
  for (int i = 0; i < 4; ++i) {
    int m = tid + 256 * i;
    float key = h[m];                       // keys = hn[:, :MS]
    float dot = key * msum[(long)b * MS_ + m];
    float xn  = sqrtMD * fabsf(key);
    float yn  = sqrtf(msq[(long)b * MS_ + m]);
    float cs  = dot / (fmaxf(xn, EPS_) * fmaxf(yn, EPS_));
    s[i] = kstr * cs;
    mx = fmaxf(mx, s[i]);
  }
  red[tid] = mx; __syncthreads();
  for (int off = 128; off > 0; off >>= 1) {
    if (tid < off) red[tid] = fmaxf(red[tid], red[tid + off]);
    __syncthreads();
  }
  mx = red[0]; __syncthreads();

  float e[4], sum = 0.f;
  #pragma unroll
  for (int i = 0; i < 4; ++i) { e[i] = __expf(s[i] - mx); sum += e[i]; }
  red[tid] = sum; __syncthreads();
  for (int off = 128; off > 0; off >>= 1) {
    if (tid < off) red[tid] += red[tid + off];
    __syncthreads();
  }
  const float inv = 1.0f / red[0];

  #pragma unroll
  for (int i = 0; i < 4; ++i) {
    int m = tid + 256 * i;
    wout[(long)b * MS_ + m] =
        gate * e[i] * inv + (1.0f - gate) * heads[(long)b * MS_ + m];
  }
}

// ====================================================================
// Fused: read_in[b,d] = sum_m read_w[b,m]*mem[b,m,d]  (on OLD memory)
//        mem_out = mem*(1 - w*erase) + w*add          (single mem pass)
// ====================================================================
__global__ void __launch_bounds__(256)
mem_update_read_kernel(const float* __restrict__ mem_in,
                       const float* __restrict__ whn,     // [B, CH]
                       const float* __restrict__ write_w, // [B, MS]
                       const float* __restrict__ read_w,  // [B, MS]
                       float* __restrict__ mem_out,       // [B, MS, MD]
                       float* __restrict__ read_in)       // [B, MD]
{
  const int b  = blockIdx.x;
  const int d  = threadIdx.x & 63;
  const int mg = threadIdx.x >> 6;     // 0..3
  float racc = 0.f;
  for (int m = mg; m < MS_; m += 4) {
    float w  = write_w[(long)b * MS_ + m];
    float er = sigm(whn[(long)b * CH_ + (MD_ + 4) + m]);       // erase
    float ad = sigm(whn[(long)b * CH_ + (2 * MD_ + 4) + m]);   // add
    float rw = read_w[(long)b * MS_ + m];
    long idx = ((long)b * MS_ + m) * MD_ + d;
    float old = mem_in[idx];
    mem_out[idx] = old * (1.0f - w * er) + w * ad;
    racc += rw * old;
  }
  __shared__ float sm[256];
  sm[threadIdx.x] = racc;
  __syncthreads();
  if (threadIdx.x < 64)
    read_in[(long)b * MD_ + threadIdx.x] =
        sm[threadIdx.x] + sm[threadIdx.x + 64] +
        sm[threadIdx.x + 128] + sm[threadIdx.x + 192];
}

__global__ void add_kernel(const float* __restrict__ a,
                           const float* __restrict__ b,
                           float* __restrict__ o, long n)
{
  long i = (long)blockIdx.x * blockDim.x + threadIdx.x;
  if (i < n) o[i] = a[i] + b[i];
}

// ====================================================================
// Host launcher
// ====================================================================
static inline void launch_gemm(const float* A, int lda, const float* W, int ldw,
                               const float* bias, float* C, int M, int N, int K,
                               int beta, hipStream_t s)
{
  constexpr int NT = 4;
  int tilesM = (M + 15) / 16;
  int tilesN = (N + 15) / 16;
  int strips = (tilesN + NT - 1) / NT;
  int waves  = tilesM * strips;
  int blocks = (waves + 7) / 8;
  gemm_wmma_bf16<NT><<<blocks, 256, 0, s>>>(A, lda, W, ldw, bias, C, M, N, K, beta);
}

static inline void launch_gemm_splitk(const float* A, int lda, const float* W,
                                      int ldw, float* C, float* part,
                                      int M, int N, int K, hipStream_t s)
{
  constexpr int NT = 8;
  constexpr int SK = 32;
  int kc = ((K / SK + 31) / 32) * 32;
  int tilesM = (M + 15) / 16;
  int tilesN = (N + 15) / 16;
  int strips = (tilesN + NT - 1) / NT;
  int waves  = tilesM * strips * SK;
  int blocks = (waves + 7) / 8;
  gemm_wmma_bf16_splitk<NT><<<blocks, 256, 0, s>>>(A, lda, W, ldw, part,
                                                   M, N, K, SK, kc);
  long MN = (long)M * N;
  splitk_reduce_kernel<<<(int)((MN + 255) / 256), 256, 0, s>>>(C, part, SK, MN);
}

static inline void launch_tdm_prefetch(const float* base, int rows, int cols,
                                       int ld, hipStream_t s)
{
  int tilesX = (cols + 63) / 64;
  int tilesY = (rows + 63) / 64;
  tdm_prefetch_kernel<<<tilesX * tilesY, 32, 0, s>>>(base, rows, cols, ld);
}

extern "C" void kernel_launch(void* const* d_in, const int* in_sizes, int n_in,
                              void* d_out, int out_size, void* d_ws, size_t ws_size,
                              hipStream_t stream) {
  (void)in_sizes; (void)n_in; (void)out_size; (void)ws_size;
  const float* x      = (const float*)d_in[0];   // [B,1,IN]
  const float* h0     = (const float*)d_in[1];   // [1,B,HD]
  const float* c0     = (const float*)d_in[2];   // [1,B,HD]
  const float* memory = (const float*)d_in[3];   // [B,MS,MD]
  const float* rheads = (const float*)d_in[4];   // [B,1,MS]
  const float* wheads = (const float*)d_in[5];   // [B,1,MS]
  const float* r_h    = (const float*)d_in[6];   // [1,B,CH]
  const float* r_c    = (const float*)d_in[7];
  const float* w_h    = (const float*)d_in[8];
  const float* w_c    = (const float*)d_in[9];
  const float* Wt     = (const float*)d_in[10];  // [HD,IN]
  const float* bt     = (const float*)d_in[11];
  const float* Wrp    = (const float*)d_in[12];  // [CD, HD+MS*MD]
  const float* brp    = (const float*)d_in[13];
  const float* Wwp    = (const float*)d_in[14];
  const float* bwp    = (const float*)d_in[15];
  const float* rWih   = (const float*)d_in[16];  // [4CH, CD]
  const float* rWhh   = (const float*)d_in[17];  // [4CH, CH]
  const float* rbih   = (const float*)d_in[18];
  const float* rbhh   = (const float*)d_in[19];
  const float* wWih   = (const float*)d_in[20];
  const float* wWhh   = (const float*)d_in[21];
  const float* wbih   = (const float*)d_in[22];
  const float* wbhh   = (const float*)d_in[23];
  const float* Wih    = (const float*)d_in[24];  // [4HD, HD+MD]
  const float* Whh    = (const float*)d_in[25];  // [4HD, HD]
  const float* bih    = (const float*)d_in[26];
  const float* bhh    = (const float*)d_in[27];

  // ---- output segments (flat concat: output, h1, c1, memory) ----
  float* o_output = (float*)d_out;
  float* o_h1  = o_output + (long)B_ * HD_;
  float* o_c1  = o_h1     + (long)B_ * HD_;
  float* o_mem = o_c1     + (long)B_ * HD_;

  // ---- workspace layout (floats) ----
  float* ws       = (float*)d_ws;
  float* w_out    = ws;                              // [B,HD]   (== xres)
  float* w_rin    = w_out    + (long)B_ * HD_;       // [B,CD]
  float* w_win    = w_rin    + (long)B_ * CD_;       // [B,CD]
  float* w_rg     = w_win    + (long)B_ * CD_;       // [B,4CH]
  float* w_wg     = w_rg     + (long)B_ * CH4_;      // [B,4CH]
  float* w_rhn    = w_wg     + (long)B_ * CH4_;      // [B,CH]
  float* w_rcn    = w_rhn    + (long)B_ * CH_;
  float* w_whn    = w_rcn    + (long)B_ * CH_;
  float* w_wcn    = w_whn    + (long)B_ * CH_;
  float* w_msum   = w_wcn    + (long)B_ * CH_;       // [B,MS]
  float* w_msq    = w_msum   + (long)B_ * MS_;
  float* w_readw  = w_msq    + (long)B_ * MS_;
  float* w_writew = w_readw  + (long)B_ * MS_;
  float* w_readin = w_writew + (long)B_ * MS_;       // [B,MD]
  float* w_g2     = w_readin + (long)B_ * MD_;       // [B,4HD]
  float* w_part   = w_g2     + (long)B_ * HD4_;      // [32,B,CD] split-K partials

  // 0) TDM: stream the two 34MB projection weights through L2 ahead of use
  launch_tdm_prefetch(Wrp, CD_, HD_ + MS_ * MD_, HD_ + MS_ * MD_, stream);
  launch_tdm_prefetch(Wwp, CD_, HD_ + MS_ * MD_, HD_ + MS_ * MD_, stream);

  // 1) out = x @ Wt.T + bt   (also serves as xres)
  launch_gemm(x, IN_, Wt, IN_, bt, w_out, B_, HD_, IN_, 0, stream);

  // 2) controller projections: ctrl_in = [h0 | memory] fused as split-K
  launch_gemm(h0, HD_, Wrp, HD_ + MS_ * MD_, brp, w_rin, B_, CD_, HD_, 0, stream);
  launch_gemm_splitk(memory, MS_ * MD_, Wrp + HD_, HD_ + MS_ * MD_,
                     w_rin, w_part, B_, CD_, MS_ * MD_, stream);
  launch_gemm(h0, HD_, Wwp, HD_ + MS_ * MD_, bwp, w_win, B_, CD_, HD_, 0, stream);
  launch_gemm_splitk(memory, MS_ * MD_, Wwp + HD_, HD_ + MS_ * MD_,
                     w_win, w_part, B_, CD_, MS_ * MD_, stream);

  // 3) controller LSTM gate GEMMs (accumulate both matmuls + both biases)
  launch_gemm(w_rin, CD_, rWih, CD_, rbih, w_rg, B_, CH4_, CD_, 0, stream);
  launch_gemm(r_h,   CH_, rWhh, CH_, rbhh, w_rg, B_, CH4_, CH_, 1, stream);
  launch_gemm(w_win, CD_, wWih, CD_, wbih, w_wg, B_, CH4_, CD_, 0, stream);
  launch_gemm(w_h,   CH_, wWhh, CH_, wbhh, w_wg, B_, CH4_, CH_, 1, stream);

  // 4) controller LSTM pointwise
  {
    long n = (long)B_ * CH_;
    int blocks = (int)((n + 255) / 256);
    lstm_pointwise_kernel<<<blocks, 256, 0, stream>>>(w_rg, r_c, w_rhn, w_rcn, B_, CH_);
    lstm_pointwise_kernel<<<blocks, 256, 0, stream>>>(w_wg, w_c, w_whn, w_wcn, B_, CH_);
  }

  // 5) memory row stats
  rowstats_kernel<<<(B_ * MS_) / 256, 256, 0, stream>>>(memory, w_msum, w_msq);

  // 6) content addressing (read + write)
  addressing_kernel<<<B_, 256, 0, stream>>>(w_rhn, rheads, w_msum, w_msq, w_readw);
  addressing_kernel<<<B_, 256, 0, stream>>>(w_whn, wheads, w_msum, w_msq, w_writew);

  // 7) fused read-einsum + memory erase/add update (single memory pass)
  mem_update_read_kernel<<<B_, 256, 0, stream>>>(memory, w_whn, w_writew, w_readw,
                                                 o_mem, w_readin);

  // 8) final LSTM gates: [out | read_in] @ Wih.T + bih + h0 @ Whh.T + bhh
  launch_gemm(w_out,    HD_, Wih,        HD_ + MD_, bih,     w_g2, B_, HD4_, HD_, 0, stream);
  launch_gemm(w_readin, MD_, Wih + HD_,  HD_ + MD_, nullptr, w_g2, B_, HD4_, MD_, 1, stream);
  launch_gemm(h0,       HD_, Whh,        HD_,       bhh,     w_g2, B_, HD4_, HD_, 1, stream);

  // 9) final LSTM pointwise -> h1, c1 directly into d_out segments
  lstm_pointwise_kernel<<<(B_ * HD_) / 256, 256, 0, stream>>>(w_g2, c0, o_h1, o_c1, B_, HD_);

  // 10) output = h1 + xres
  add_kernel<<<(B_ * HD_) / 256, 256, 0, stream>>>(o_h1, w_out, o_output, (long)B_ * HD_);
}
